// Linear4bit_25486335934617
// MI455X (gfx1250) — matrix operand order repaired and checked by reading, not compiled, not run on hardware
//
#include <hip/hip_runtime.h>
#include <stdint.h>

#define M_DIM 8192
#define K_DIM 4096
#define N_DIM 4096

#define BM 128
#define BN 128
#define BK 32
#define KTILES (K_DIM / BK)   // 128

typedef __attribute__((ext_vector_type(16))) _Float16 v16h;
typedef __attribute__((ext_vector_type(8)))  _Float16 v8h;
typedef __attribute__((ext_vector_type(8)))  float    v8f;
typedef __attribute__((ext_vector_type(4)))  float    v4f;
typedef __attribute__((ext_vector_type(4)))  int      v4i;

union F16x16 { v16h v; v8h h[2]; };

// ---------------- CDNA5 async global->LDS path (probe-guarded) ----------------
#if defined(__gfx1250__)
#if __has_builtin(__builtin_amdgcn_global_load_async_to_lds_b128)
#define ASYNC_LDS 1
#endif
#endif

__device__ __forceinline__ void copy16_g2l(const _Float16* g, _Float16* l) {
#ifdef ASYNC_LDS
  // clang-22 prototype (from diagnostic): generic v4i* params, 2 imm ints.
  __builtin_amdgcn_global_load_async_to_lds_b128((v4i*)(void*)g, (v4i*)l, 0, 0);
#else
  *(v8h*)l = *(const v8h*)g;   // global_load_b128 + ds_store_b128 fallback
#endif
}

__device__ __forceinline__ void wait_async_copies() {
#ifdef ASYNC_LDS
#if __has_builtin(__builtin_amdgcn_s_wait_asynccnt)
  __builtin_amdgcn_s_wait_asynccnt(0);
#else
  asm volatile("s_wait_asynccnt 0" ::: "memory");
#endif
#endif
}

// ---------------- Pre-pass 1: x fp32 -> f16 ----------------
__global__ __launch_bounds__(256) void cvt_x_f16(const float* __restrict__ x,
                                                 _Float16* __restrict__ xf) {
  const int i = (blockIdx.x * 256 + threadIdx.x) * 8;
  v4f a = *(const v4f*)(x + i);
  v4f b = *(const v4f*)(x + i + 4);
  v8h o;
  o[0] = (_Float16)a[0]; o[1] = (_Float16)a[1];
  o[2] = (_Float16)a[2]; o[3] = (_Float16)a[3];
  o[4] = (_Float16)b[0]; o[5] = (_Float16)b[1];
  o[6] = (_Float16)b[2]; o[7] = (_Float16)b[3];
  *(v8h*)(xf + i) = o;
}

// ---------------- Pre-pass 2: int4 dequant -> f16 [N,K] row-major ----------------
// Each packed int32 holds one byte: low nibble -> even k, high nibble -> odd k.
// Offset-binary: signed value = nibble - 8 (equivalent to ref's XOR 0x88 + sext).
__global__ __launch_bounds__(256) void dequant_w4_f16(const int* __restrict__ wp,
                                                      const float* __restrict__ scales,
                                                      _Float16* __restrict__ wf) {
  const int t = blockIdx.x * 256 + threadIdx.x;
  const int base = t * 4;                       // 4 packed bytes -> 8 halves
  const int row = base / (K_DIM / 2);           // all 4 in same row (2048 % 4 == 0)
  const float s = scales[row];
  int4 p = *(const int4*)(wp + base);
  int pk[4] = {p.x, p.y, p.z, p.w};
  v8h o;
#pragma unroll
  for (int j = 0; j < 4; ++j) {
    int lo = (pk[j] & 0xF) - 8;
    int hi = ((pk[j] >> 4) & 0xF) - 8;
    o[2 * j + 0] = (_Float16)((float)lo * s);
    o[2 * j + 1] = (_Float16)((float)hi * s);
  }
  *(v8h*)(wf + (size_t)base * 2) = o;
}

// ---------------- LDS fragment-major swizzle ----------------
// A tile: 8 m-tiles of 16x32 halves; lane L owns 16B chunks j = L/16 and j = 2 + L/16.
// idx(m, j) with j = 16B chunk (8 halves) along K:
__device__ __forceinline__ int a_swz(int m, int j) {
  return (m >> 4) * 512 + ((j >> 1) << 8) + ((((j & 1) << 4) + (m & 15)) << 3);
}
__device__ __forceinline__ int b_swz(int n, int j) {
  return (n >> 4) * 512 + ((j >> 1) << 8) + ((((j & 1) << 4) + (n & 15)) << 3);
}

// ---------------- Main GEMM: C = A(f16)[M,K] * B(f16)[N,K]^T + bias ----------------
__global__ __launch_bounds__(256) void gemm_w4a16_wmma(
    const _Float16* __restrict__ A,   // x_f16 [M,K]
    const _Float16* __restrict__ B,   // w_f16 [N,K]
    const float* __restrict__ bias,   // [N]
    float* __restrict__ C) {          // [M,N]
  __shared__ _Float16 lA[2][BM * BK];   // 8 KB each buffer
  __shared__ _Float16 lB[2][BN * BK];

  const int tid  = threadIdx.x;
  const int lane = tid & 31;
  const int wave = tid >> 5;        // 0..7
  const int wm   = wave >> 2;       // 0..1  (64 rows each)
  const int wn   = wave & 3;        // 0..3  (32 cols each)
  const int m0   = blockIdx.y * BM;
  const int n0   = blockIdx.x * BN;
  const int l16  = lane & 15;

  // Stage one BK-slice of A and B into LDS buffer `buf` (async when available).
  auto stage = [&](int buf, int kt) {
    const _Float16* ga = A + (size_t)m0 * K_DIM + kt * BK;
    const _Float16* gb = B + (size_t)n0 * K_DIM + kt * BK;
#pragma unroll
    for (int i = 0; i < 2; ++i) {
      const int c   = tid + i * 256;   // 512 16B chunks per tile
      const int row = c >> 2;
      const int j   = c & 3;           // K-chunk within row
      copy16_g2l(ga + (size_t)row * K_DIM + j * 8, &lA[buf][a_swz(row, j)]);
      copy16_g2l(gb + (size_t)row * K_DIM + j * 8, &lB[buf][b_swz(row, j)]);
    }
  };

  v8f acc[4][2];
#pragma unroll
  for (int tm = 0; tm < 4; ++tm)
#pragma unroll
    for (int tn = 0; tn < 2; ++tn) acc[tm][tn] = (v8f)(0.0f);

  stage(0, 0);
  wait_async_copies();
  __syncthreads();

  for (int kt = 0; kt < KTILES; ++kt) {
    const int cur = kt & 1;
    if (kt + 1 < KTILES) stage(cur ^ 1, kt + 1);

    if (kt + 2 < KTILES) {  // global_prefetch_b8 two tiles ahead
      const int row = tid >> 1, half = (tid & 1) * 16;
      __builtin_prefetch(A + (size_t)(m0 + row) * K_DIM + (kt + 2) * BK + half, 0, 0);
      __builtin_prefetch(B + (size_t)(n0 + row) * K_DIM + (kt + 2) * BK + half, 0, 0);
    }

    // Load fragments: wave-contiguous ds_load_b128 pairs.
    F16x16 af[4];
#pragma unroll
    for (int tm = 0; tm < 4; ++tm) {
      const int tA = wm * 4 + tm;     // m-tile index 0..7
      af[tm].h[0] = *(const v8h*)&lA[cur][tA * 512 +       lane * 8];
      af[tm].h[1] = *(const v8h*)&lA[cur][tA * 512 + 256 + lane * 8];
    }
    F16x16 bf[2];
#pragma unroll
    for (int tn = 0; tn < 2; ++tn) {
      const int tB = wn * 2 + tn;     // n-tile index 0..7
      bf[tn].h[0] = *(const v8h*)&lB[cur][tB * 512 +       lane * 8];
      bf[tn].h[1] = *(const v8h*)&lB[cur][tB * 512 + 256 + lane * 8];
    }

#pragma unroll
    for (int tm = 0; tm < 4; ++tm)
#pragma unroll
      for (int tn = 0; tn < 2; ++tn)
        acc[tm][tn] = __builtin_amdgcn_wmma_f32_16x16x32_f16(
            false, af[tm].v, false, bf[tn].v,
            (short)0, acc[tm][tn], false, false);

    wait_async_copies();
    __syncthreads();
  }

  // Epilogue: fused bias + f32 store.
  const int half8 = (lane >> 4) * 8;
#pragma unroll
  for (int tn = 0; tn < 2; ++tn) {
    const int n = n0 + wn * 32 + tn * 16 + l16;
    const float bv = bias[n];
#pragma unroll
    for (int tm = 0; tm < 4; ++tm) {
      const int mb = m0 + wm * 64 + tm * 16 + half8;
#pragma unroll
      for (int r = 0; r < 8; ++r)
        C[(size_t)(mb + r) * N_DIM + n] = acc[tm][tn][r] + bv;
    }
  }
}

// ---------------- Host launcher ----------------
extern "C" void kernel_launch(void* const* d_in, const int* in_sizes, int n_in,
                              void* d_out, int out_size, void* d_ws, size_t ws_size,
                              hipStream_t stream) {
  const float* x     = (const float*)d_in[0];
  const int*   wp    = (const int*)d_in[1];
  const float* scale = (const float*)d_in[2];
  const float* bias  = (const float*)d_in[3];
  float* out = (float*)d_out;

  _Float16* xf = (_Float16*)d_ws;                                   // 64 MB
  _Float16* wf = (_Float16*)((char*)d_ws + (size_t)M_DIM * K_DIM * 2);  // +32 MB

  cvt_x_f16<<<(M_DIM * K_DIM) / (256 * 8), 256, 0, stream>>>(x, xf);
  dequant_w4_f16<<<(N_DIM * (K_DIM / 2)) / (256 * 4), 256, 0, stream>>>(wp, scale, wf);

  dim3 grid(N_DIM / BN, M_DIM / BM);  // 32 x 64
  gemm_w4a16_wmma<<<grid, 256, 0, stream>>>(xf, wf, bias, out);
}
